// GroupAttentionLayer_65343632441906
// MI455X (gfx1250) — compile-verified, matrix operand order checked by reading, not run on hardware
//
#include <hip/hip_runtime.h>

// ---------------- CDNA5 WMMA types ----------------
typedef __attribute__((ext_vector_type(16))) _Float16 v16h;
typedef __attribute__((ext_vector_type(8)))  _Float16 v8h;
typedef __attribute__((ext_vector_type(8)))  float    v8f;

#define ALPHA 0.3f
#define BN_EPS 1e-3f
#define CCH   256          // channels
#define NB    4            // batch
#define HWSZ  4096         // 64*64 spatial per batch
#define NROWS 16384        // NB*HWSZ
#define NWIN  16           // windows per batch
#define WINP  256          // positions per window (16x16)

// gfx1250 async global->LDS copy (ASYNCcnt-tracked), guarded so we always compile
#if defined(__has_builtin)
#if __has_builtin(__builtin_amdgcn_global_load_async_to_lds_b128) && \
    __has_builtin(__builtin_amdgcn_s_wait_asynccnt)
#define USE_ASYNC_LDS 1
#endif
#endif

#ifdef USE_ASYNC_LDS
// builtin signature (from hipcc diagnostic): param0 = v4i __device__* (AS1),
// param1 = LDS dest, then i32 offset, i32 cpol
typedef int v4i __attribute__((vector_size(16)));
typedef __attribute__((address_space(1))) v4i* gas_t;
typedef __attribute__((address_space(3))) v4i* las_t;
#endif

union FragH { v16h v; v8h h[2]; };

// Load an f16 16x32 A-fragment (or 32x16 B-fragment, its mirror) from a
// row-major [rows][ld] f16 matrix in GLOBAL memory. ISA layout: lanes 0-15 ->
// row, K chunks {0..7,16..23}; lanes 16-31 -> row, K chunks {8..15,24..31}.
__device__ __forceinline__ FragH load_frag(const _Float16* __restrict__ base,
                                           int ld, int row0, int k0) {
  int lane = threadIdx.x & 31;
  int r    = lane & 15;
  int kb   = k0 + ((lane & 16) ? 8 : 0);
  const _Float16* p = base + (size_t)(row0 + r) * ld + kb;
  FragH f;
  f.h[0] = *(const v8h*)(p);
  f.h[1] = *(const v8h*)(p + 16);
  return f;
}

// Store a 16x16 f32 C/D fragment to a row-major [rows][ld] f32 matrix.
// D layout: VGPR r -> M = r + (lane<16?0:8), N = lane&15.
__device__ __forceinline__ void store_tile(float* __restrict__ dst, int ld,
                                           int m0, int n0, v8f acc) {
  int lane = threadIdx.x & 31;
  int n    = n0 + (lane & 15);
  int mb   = m0 + ((lane & 16) ? 8 : 0);
#pragma unroll
  for (int r = 0; r < 8; ++r) dst[(size_t)(mb + r) * ld + n] = acc[r];
}

// Cooperative stage of a [rows x colh] f16 sub-matrix (row stride gld halves in
// global) into a dense LDS buffer, 16B granules. Async on gfx1250.
__device__ __forceinline__ void stage2d(const _Float16* __restrict__ g, int gld,
                                        _Float16* l, int rows, int colh) {
  int tpr   = colh >> 3;          // 16-byte transfers per row
  int total = rows * tpr;
  for (int tt = threadIdx.x; tt < total; tt += (int)blockDim.x) {
    int rr = tt / tpr, seg = tt - rr * tpr;
    const _Float16* gp = g + (size_t)rr * gld + seg * 8;
    _Float16*       lp = l + (size_t)tt * 8;
#ifdef USE_ASYNC_LDS
    __builtin_amdgcn_global_load_async_to_lds_b128(
        (gas_t)(uintptr_t)gp, (las_t)(uint32_t)(uintptr_t)lp, 0, 0);
#else
    *(v8h*)lp = *(const v8h*)gp;
#endif
  }
}

__device__ __forceinline__ void stage_wait() {
#ifdef USE_ASYNC_LDS
  __builtin_amdgcn_s_wait_asynccnt(0);
#endif
  __syncthreads();
}

// ---------------- pipeline kernels ----------------

__global__ void k_cvt_x(const float* __restrict__ X, _Float16* __restrict__ Xh) {
  size_t i = (size_t)blockIdx.x * 256 + threadIdx.x;
  Xh[i] = (_Float16)X[i];
}

// weights: store transposed (out-channel major) so B-fragments load contiguously
__global__ void k_cvt_w(const float* __restrict__ Wq, const float* __restrict__ Wk,
                        const float* __restrict__ Wv, _Float16* __restrict__ Wht) {
  int t = blockIdx.x >> 8;    // 0..2
  int d = blockIdx.x & 255;   // out channel
  int c = threadIdx.x;        // in channel
  const float* W = (t == 0) ? Wq : (t == 1) ? Wk : Wv;
  Wht[(size_t)t * 65536 + (size_t)d * 256 + c] = (_Float16)W[(size_t)c * 256 + d];
}

// Y[16384,256] = Xh[16384,256] @ W  (W given transposed: Wt[d][c])
__global__ void __launch_bounds__(256) k_gemm(const _Float16* __restrict__ Xh,
                                              const _Float16* __restrict__ Wt,
                                              float* __restrict__ Y) {
  int w  = threadIdx.x >> 5;
  int m0 = blockIdx.x * 128 + w * 16;
  int c0 = blockIdx.y * 64;
  v8f acc[4] = {};
#pragma unroll
  for (int kk = 0; kk < 8; ++kk) {
    FragH a = load_frag(Xh, CCH, m0, kk * 32);
#pragma unroll
    for (int j = 0; j < 4; ++j) {
      FragH b = load_frag(Wt, CCH, c0 + j * 16, kk * 32);
      acc[j] = __builtin_amdgcn_wmma_f32_16x16x32_f16(
          false, a.v, false, b.v, (short)0, acc[j], false, false);
    }
  }
#pragma unroll
  for (int j = 0; j < 4; ++j) store_tile(Y, CCH, m0, c0 + j * 16, acc[j]);
}

// per-channel sum / sum-of-squares (BN training stats), block = 64 rows
__global__ void k_colstats(const float* __restrict__ Y, float* __restrict__ sum,
                           float* __restrict__ sumsq) {
  int c  = threadIdx.x;
  int r0 = blockIdx.x * 64;
  float s = 0.f, s2 = 0.f;
#pragma unroll 4
  for (int i = 0; i < 64; ++i) {
    float v = Y[(size_t)(r0 + i) * CCH + c];
    s += v; s2 += v * v;
  }
  atomicAdd(&sum[c], s);
  atomicAdd(&sumsq[c], s2);
}

// BN + leaky + f16 convert; scatter to GEMM-friendly layouts.
// mode 0: Qh[row][c]   mode 1: Kw[b][n][p][c]   mode 2: Vwt[b][n][c][p]
__global__ void k_bn_apply(const float* __restrict__ Y, const float* __restrict__ sum,
                           const float* __restrict__ sumsq, const float* __restrict__ gamma,
                           const float* __restrict__ beta, _Float16* __restrict__ Qh,
                           _Float16* __restrict__ Kw, _Float16* __restrict__ Vwt, int mode) {
  int row = blockIdx.x;
  int c   = threadIdx.x;
  float mu  = sum[c] * (1.f / (float)NROWS);
  float var = sumsq[c] * (1.f / (float)NROWS) - mu * mu;
  float sc  = gamma[c] * rsqrtf(var + BN_EPS);
  float sh  = beta[c] - mu * sc;
  float v = Y[(size_t)row * CCH + c] * sc + sh;
  v = (v > 0.f) ? v : ALPHA * v;
  _Float16 h = (_Float16)v;
  if (mode == 0) {
    Qh[(size_t)row * CCH + c] = h;
  } else {
    int b = row >> 12, rr = row & 4095;
    int hh = rr >> 6, ww = rr & 63;
    int n  = (hh >> 4) * 4 + (ww >> 4);
    int p  = (hh & 15) * 16 + (ww & 15);
    size_t wb = (size_t)(b * NWIN + n) * 65536;
    if (mode == 1) Kw[wb + (size_t)p * CCH + c] = h;
    else           Vwt[wb + (size_t)c * WINP + p] = h;
  }
}

// Fused attention. Per wave: 16-row Q strip resident as 8 A-fragments; per
// window:
//   phase 1: S = leaky(Q Kn^T / 16), K j-tiles async-staged to LDS (shared by
//            all 4 waves), S strip written to LDS in D-layout (f16)
//   phase 2: yout += S @ Vn, V k-chunks async-staged to LDS, S re-read from
//            LDS in A-layout
__global__ void __launch_bounds__(128) k_attn(const _Float16* __restrict__ Qh,
                                              const _Float16* __restrict__ Kw,
                                              const _Float16* __restrict__ Vwt,
                                              float* __restrict__ yout) {
  __shared__ __align__(16) _Float16 Ss[4][16][272];  // 34,816 B (544B row stride)
  __shared__ __align__(16) _Float16 Kst[16][256];    //  8,192 B: one K j-tile
  __shared__ __align__(16) _Float16 Vst[256][32];    // 16,384 B: one V k-chunk
  int w    = threadIdx.x >> 5;
  int lane = threadIdx.x & 31;
  int b    = blockIdx.y;
  int m0   = blockIdx.x * 64 + w * 16;
  const _Float16* Qb = Qh + (size_t)b * HWSZ * CCH;

  // cache the 16x256 Q strip as 8 A-fragments (reused 16 windows x 16 tiles)
  FragH qf[8];
#pragma unroll
  for (int kk = 0; kk < 8; ++kk) qf[kk] = load_frag(Qb, CCH, m0, kk * 32);

  v8f acc2[16];
#pragma unroll
  for (int j = 0; j < 16; ++j) acc2[j] = {};

  for (int n = 0; n < NWIN; ++n) {
    const _Float16* Kb = Kw  + (size_t)(b * NWIN + n) * 65536;
    const _Float16* Vb = Vwt + (size_t)(b * NWIN + n) * 65536;
    if (n + 1 < NWIN) {   // pull next window toward L2
      __builtin_prefetch(Kb + 65536, 0, 1);
      __builtin_prefetch(Vb + 65536, 0, 1);
    }
    // ---- phase 1: S strip ----
    for (int j = 0; j < 16; ++j) {
      // stage K rows [j*16, j*16+16) x all 256 ch (contiguous 8KB) into LDS
      stage2d(Kb + (size_t)j * 16 * CCH, CCH, &Kst[0][0], 16, 256);
      stage_wait();
      v8f s = {};
#pragma unroll
      for (int kk = 0; kk < 8; ++kk) {
        int kb = kk * 32 + ((lane & 16) ? 8 : 0);
        FragH bf;
        bf.h[0] = *(const v8h*)&Kst[lane & 15][kb];       // ds_load_b128
        bf.h[1] = *(const v8h*)&Kst[lane & 15][kb + 16];
        s = __builtin_amdgcn_wmma_f32_16x16x32_f16(
            false, qf[kk].v, false, bf.v, (short)0, s, false, false);
      }
      int nn = j * 16 + (lane & 15);
      int mb = (lane & 16) ? 8 : 0;
#pragma unroll
      for (int r = 0; r < 8; ++r) {
        float v = s[r] * 0.0625f;              // 1/sqrt(256)
        v = (v > 0.f) ? v : ALPHA * v;
        Ss[w][mb + r][nn] = (_Float16)v;
      }
      __syncthreads();                          // Kst consumed; Ss tile done
    }
    // ---- phase 2: yout += S @ V ----
    for (int kk = 0; kk < 8; ++kk) {
      // stage V chunk: all 256 ch x positions [kk*32, kk*32+32) (16KB)
      stage2d(Vb + kk * 32, WINP, &Vst[0][0], 256, 32);
      stage_wait();
      FragH a2;
      int r  = lane & 15;
      int kb = kk * 32 + ((lane & 16) ? 8 : 0);
      a2.h[0] = *(const v8h*)&Ss[w][r][kb];                // ds_load_b128
      a2.h[1] = *(const v8h*)&Ss[w][r][kb + 16];
#pragma unroll
      for (int j = 0; j < 16; ++j) {
        int kb2 = (lane & 16) ? 8 : 0;
        FragH bf;
        bf.h[0] = *(const v8h*)&Vst[j * 16 + (lane & 15)][kb2];
        bf.h[1] = *(const v8h*)&Vst[j * 16 + (lane & 15)][kb2 + 16];
        acc2[j] = __builtin_amdgcn_wmma_f32_16x16x32_f16(
            false, a2.v, false, bf.v, (short)0, acc2[j], false, false);
      }
      __syncthreads();                          // Vst consumed
    }
  }
  float* Yo = yout + (size_t)b * HWSZ * CCH;
#pragma unroll
  for (int j = 0; j < 16; ++j) store_tile(Yo, CCH, m0, j * 16, acc2[j]);
}

// BN(yout) then softmax jointly over the 4096 spatial positions, per (b, c)
__global__ void k_softmax(const float* __restrict__ yout, const float* __restrict__ sum1,
                          const float* __restrict__ sumsq1, const float* __restrict__ g1,
                          const float* __restrict__ b1, float* __restrict__ out) {
  __shared__ float red[256];
  int b = blockIdx.x >> 8;
  int c = blockIdx.x & 255;
  int t = threadIdx.x;
  float mu  = sum1[c] * (1.f / (float)NROWS);
  float var = sumsq1[c] * (1.f / (float)NROWS) - mu * mu;
  float sc  = g1[c] * rsqrtf(var + BN_EPS);
  float sh  = b1[c] - mu * sc;
  const float* Yb = yout + (size_t)b * HWSZ * CCH;

  float m = -1e30f;
  for (int r = t; r < HWSZ; r += 256)
    m = fmaxf(m, Yb[(size_t)r * CCH + c] * sc + sh);
  red[t] = m; __syncthreads();
  for (int s = 128; s > 0; s >>= 1) {
    if (t < s) red[t] = fmaxf(red[t], red[t + s]);
    __syncthreads();
  }
  m = red[0]; __syncthreads();

  float se = 0.f;
  for (int r = t; r < HWSZ; r += 256)
    se += __expf(Yb[(size_t)r * CCH + c] * sc + sh - m);
  red[t] = se; __syncthreads();
  for (int s = 128; s > 0; s >>= 1) {
    if (t < s) red[t] += red[t + s];
    __syncthreads();
  }
  float inv = 1.f / red[0];

  float* Ob = out + (size_t)b * HWSZ * CCH;
  for (int r = t; r < HWSZ; r += 256) {
    float v = Yb[(size_t)r * CCH + c] * sc + sh;
    Ob[(size_t)r * CCH + c] = __expf(v - m) * inv;
  }
}

// ---------------- workspace layout (bytes) ----------------
#define OFF_XH    ((size_t)0)          //  8 MiB  f16 [16384][256]
#define OFF_WHT   ((size_t)8388608)    //  384 KiB f16 [3][256][256] (transposed)
#define OFF_QH    ((size_t)8781824)    //  8 MiB  f16 [4][4096][256]
#define OFF_KW    ((size_t)17170432)   //  8 MiB  f16 [4][16][256 p][256 c]
#define OFF_VWT   ((size_t)25559040)   //  8 MiB  f16 [4][16][256 c][256 p]
#define OFF_YTMP  ((size_t)33947648)   // 16 MiB  f32 [16384][256]
#define OFF_YOUT  ((size_t)50724864)   // 16 MiB  f32 [16384][256]
#define OFF_STATS ((size_t)67502080)   //  8 KiB  8 x 256 f32

extern "C" void kernel_launch(void* const* d_in, const int* in_sizes, int n_in,
                              void* d_out, int out_size, void* d_ws, size_t ws_size,
                              hipStream_t stream) {
  const float* X  = (const float*)d_in[0];
  const float* Wq = (const float*)d_in[1];
  const float* gq = (const float*)d_in[2];
  const float* bq = (const float*)d_in[3];
  const float* Wk = (const float*)d_in[4];
  const float* gk = (const float*)d_in[5];
  const float* bk = (const float*)d_in[6];
  const float* Wv = (const float*)d_in[7];
  const float* gv = (const float*)d_in[8];
  const float* bv = (const float*)d_in[9];
  const float* g1 = (const float*)d_in[10];
  const float* b1 = (const float*)d_in[11];
  float* out = (float*)d_out;
  char*  ws  = (char*)d_ws;

  _Float16* Xh  = (_Float16*)(ws + OFF_XH);
  _Float16* Wht = (_Float16*)(ws + OFF_WHT);
  _Float16* Qh  = (_Float16*)(ws + OFF_QH);
  _Float16* Kw  = (_Float16*)(ws + OFF_KW);
  _Float16* Vwt = (_Float16*)(ws + OFF_VWT);
  float* Ytmp = (float*)(ws + OFF_YTMP);
  float* Yout = (float*)(ws + OFF_YOUT);
  float* st   = (float*)(ws + OFF_STATS);
  float *sQ = st, *qQ = st + 256, *sK = st + 512, *qK = st + 768;
  float *sV = st + 1024, *qV = st + 1280, *s1 = st + 1536, *q1 = st + 1792;

  (void)hipMemsetAsync(ws + OFF_STATS, 0, 8192, stream);

  k_cvt_x<<<NROWS, 256, 0, stream>>>(X, Xh);
  k_cvt_w<<<768, 256, 0, stream>>>(Wq, Wk, Wv, Wht);

  // Q
  k_gemm<<<dim3(128, 4), 256, 0, stream>>>(Xh, Wht, Ytmp);
  k_colstats<<<256, 256, 0, stream>>>(Ytmp, sQ, qQ);
  k_bn_apply<<<NROWS, 256, 0, stream>>>(Ytmp, sQ, qQ, gq, bq, Qh, Kw, Vwt, 0);
  // K
  k_gemm<<<dim3(128, 4), 256, 0, stream>>>(Xh, Wht + 65536, Ytmp);
  k_colstats<<<256, 256, 0, stream>>>(Ytmp, sK, qK);
  k_bn_apply<<<NROWS, 256, 0, stream>>>(Ytmp, sK, qK, gk, bk, Qh, Kw, Vwt, 1);
  // V
  k_gemm<<<dim3(128, 4), 256, 0, stream>>>(Xh, Wht + 131072, Ytmp);
  k_colstats<<<256, 256, 0, stream>>>(Ytmp, sV, qV);
  k_bn_apply<<<NROWS, 256, 0, stream>>>(Ytmp, sV, qV, gv, bv, Qh, Kw, Vwt, 2);

  // fused window attention (bulk of the ~75 GFLOP)
  k_attn<<<dim3(64, NB), 128, 0, stream>>>(Qh, Kw, Vwt, Yout);

  // BN1 stats + softmax epilogue
  k_colstats<<<256, 256, 0, stream>>>(Yout, s1, q1);
  k_softmax<<<NB * 256, 256, 0, stream>>>(Yout, s1, q1, g1, b1, out);
}
